// MoFo_Circulant_DP_Backbone_5437428597103
// MI455X (gfx1250) — matrix operand order repaired and checked by reading, not compiled, not run on hardware
//
#include <hip/hip_runtime.h>
#include <hip/hip_bf16.h>
#include <math.h>

typedef __attribute__((ext_vector_type(16))) _Float16 v16h;
typedef __attribute__((ext_vector_type(8)))  _Float16 v8h;
typedef __attribute__((ext_vector_type(8)))  float    v8f;

#define Nn 8192
#define Dd 128
#define Bb 16
#define BL 819              // int(8192*0.1)
#define KTAP 128            // half-width of truncated circulant filter
#define SEG 512             // EMA segment; 0.9^512 ~ 4e-24 -> carry-in approx exact in f32
#define NSEG (Nn / SEG)
#define WROWS 352           // 64 outputs + 256 halo + 32 real-data pad (taps are 0 there)

// mask(f) = min(2*G(f-pf) + G(f-2pf) + G(f-3pf), 1),  G gaussian with sigma=1/48
__device__ __forceinline__ float gmask(float f) {
    const float pf = 1.0f / 24.0f;
    const float inv_sg = 48.0f;
    float d1 = (f - pf) * inv_sg;
    float d2 = (f - 2.0f * pf) * inv_sg;
    float d3 = (f - 3.0f * pf) * inv_sg;
    float m = 2.0f * __expf(-0.5f * d1 * d1)
            + __expf(-0.5f * d2 * d2)
            + __expf(-0.5f * d3 * d3);
    return fminf(m, 1.0f);
}

// grev[m], m in [0,320): reversed zero-padded taps so A[r][w] = grev[w-r+15].
__global__ void mofo_taps_kernel(float* __restrict__ grev) {
    int m = threadIdx.x;               // 320 threads
    float h = 0.0f;
    if (m >= 15 && m <= 271) {
        int lag = __builtin_abs(143 - m);   // 0..128
        float T0  = gmask(0.0f);
        float TN2 = gmask(0.5f);
        float acc = 0.0f;
        for (int f = 1; f < Nn / 2; ++f) {
            float mk = gmask((float)f * (1.0f / (float)Nn));
            int ph = (f * lag) & (Nn - 1);
            acc += mk * __cosf((float)ph * (6.283185307179586f / (float)Nn));
        }
        float s2 = (lag & 1) ? -TN2 : TN2;
        h = (T0 + s2 + 2.0f * acc) * (1.0f / (float)Nn);
    }
    grev[m] = h;
}

// EMA pass 1: per-segment endpoint with zero carry-in (error ~0.9^512, below f32 eps).
__global__ void mofo_ema_seg_kernel(const float* __restrict__ x, float* __restrict__ eend) {
    int b = blockIdx.x, s = blockIdx.y, d = threadIdx.x;
    size_t base = (((size_t)b * Nn) + (size_t)s * SEG) * Dd + d;
    float c = (s == 0) ? x[base] : 0.1f * x[base];
    for (int k = 1; k < SEG; ++k) {
        size_t idx = base + (size_t)k * Dd;
        __builtin_prefetch(&x[idx + 32 * Dd], 0, 0);   // global_prefetch_b8
        c = 0.1f * x[idx] + 0.9f * c;
    }
    eend[((size_t)b * NSEG + s) * Dd + d] = c;
}

// EMA pass 2: replay each segment from the previous segment's endpoint.
__global__ void mofo_ema_fill_kernel(const float* __restrict__ x,
                                     const float* __restrict__ eend,
                                     float* __restrict__ ema) {
    int b = blockIdx.x, s = blockIdx.y, d = threadIdx.x;
    size_t base = (((size_t)b * Nn) + (size_t)s * SEG) * Dd + d;
    float c;
    int k0;
    if (s == 0) {
        c = x[base];
        ema[base] = c;
        k0 = 1;
    } else {
        c = eend[((size_t)b * NSEG + (s - 1)) * Dd + d];
        k0 = 0;
    }
    for (int k = k0; k < SEG; ++k) {
        size_t idx = base + (size_t)k * Dd;
        c = 0.1f * x[idx] + 0.9f * c;
        ema[idx] = c;
    }
}

// Circulant FIR via WMMA: block = (64-row n-tile) x (64-wide d-half) x b.
// 128 threads = 4 waves; wave owns a d-subtile, iterates 4 n-subtiles.
__global__ void __launch_bounds__(128)
mofo_conv_kernel(const float* __restrict__ x, const float* __restrict__ grev,
                 float* __restrict__ seasonal) {
    __shared__ __align__(32) _Float16 xh[64 * WROWS];   // 45056 B, [d_local][w]
    __shared__ __align__(32) _Float16 gh[320];
    __shared__ __align__(16) float    gstage[320];      // async-copy staging
    int t = threadIdx.x;
    int n0 = blockIdx.x * 64;
    int dbase = blockIdx.y * 64;
    int b = blockIdx.z;
    const float* xb = x + (size_t)b * Nn * Dd;

    // Async copy taps to LDS (ASYNCcnt path), overlapped with window staging.
    if (t < 80) {
        unsigned ldsa = (unsigned)(size_t)(void*)&gstage[t * 4];
        unsigned voff = t * 16u;
        asm volatile("global_load_async_to_lds_b128 %0, %1, %2 offset:0"
                     :: "v"(ldsa), "v"(voff), "s"(grev) : "memory");
    }

    int dl = t & 63;
    int w0 = t >> 6;
    #pragma unroll 4
    for (int j = 0; j < WROWS / 2; ++j) {
        int w = w0 + 2 * j;
        int row = (n0 - KTAP + w) & (Nn - 1);           // circular window
        xh[dl * WROWS + w] = (_Float16)xb[(size_t)row * Dd + dbase + dl];
    }
    asm volatile("s_wait_asynccnt 0x0" ::: "memory");
    __syncthreads();
    for (int i = t; i < 320; i += 128) gh[i] = (_Float16)gstage[i];
    __syncthreads();

    int wave = t >> 5, L = t & 31;
    int r    = L & 15;                 // A's M index
    int klo  = (L >= 16) ? 8 : 0;      // unused; A fragment also uses k16
    int k16  = (L >= 16) ? 16 : 0;     // B-fragment K offset per lane-half
    int dl2  = wave * 16 + (L & 15);   // B column (d_local)
    v8f c0 = {}, c1 = {}, c2 = {}, c3 = {};
    for (int ch = 0; ch < 9; ++ch) {
        v16h a;                        // Toeplitz tap fragment, shared by 4 subtiles
        #pragma unroll
        for (int i = 0; i < 8; ++i) {
            a[i]     = gh[ch * 32 + k16 + i      - r + 15];
            a[i + 8] = gh[ch * 32 + k16 + 8 + i  - r + 15];
        }
        const _Float16* bp = &xh[dl2 * WROWS + ch * 32 + k16];
        v16h bf0 = *(const v16h*)(bp);
        v16h bf1 = *(const v16h*)(bp + 16);
        v16h bf2 = *(const v16h*)(bp + 32);
        v16h bf3 = *(const v16h*)(bp + 48);
        c0 = __builtin_amdgcn_wmma_f32_16x16x32_f16(false, a, false, bf0, (short)0, c0, false, false);
        c1 = __builtin_amdgcn_wmma_f32_16x16x32_f16(false, a, false, bf1, (short)0, c1, false, false);
        c2 = __builtin_amdgcn_wmma_f32_16x16x32_f16(false, a, false, bf2, (short)0, c2, false, false);
        c3 = __builtin_amdgcn_wmma_f32_16x16x32_f16(false, a, false, bf3, (short)0, c3, false, false);
    }
    int dcol = dbase + dl2;
    int mofs = (L >= 16) ? 8 : 0;
    #pragma unroll
    for (int v = 0; v < 8; ++v) {
        size_t rowb = (size_t)b * Nn + n0 + v + mofs;
        seasonal[(rowb +  0) * Dd + dcol] = c0[v];
        seasonal[(rowb + 16) * Dd + dcol] = c1[v];
        seasonal[(rowb + 32) * Dd + dcol] = c2[v];
        seasonal[(rowb + 48) * Dd + dcol] = c3[v];
    }
}

// trend = x - seasonal; proj = trend @ W^T + b (WMMA); blend head; RMS-norm; mix.
// block = 32-row n-tile; 8 waves own e-tiles, 2 n-subtiles each.
__global__ void __launch_bounds__(256)
mofo_fuse_kernel(const float* __restrict__ x, const float* __restrict__ W,
                 const float* __restrict__ bias, const float* __restrict__ scale,
                 const float* __restrict__ offset, const float* __restrict__ logit_alpha,
                 const float* __restrict__ seasonal, float* __restrict__ out) {
    __shared__ __align__(32) _Float16 Wh[Dd * Dd];   // 32 KB
    __shared__ __align__(32) _Float16 Th[32 * Dd];   // 8 KB trend tile
    __shared__ __align__(32) float    Sb[32 * Dd];   // 16 KB blended seasonal
    __shared__ float Rms[32];

    int t = threadIdx.x;
    int n0 = blockIdx.x * 32;
    int b = blockIdx.y;

    #pragma unroll 8
    for (int j = 0; j < (Dd * Dd) / 256; ++j) {
        int idx = t + 256 * j;
        Wh[idx] = (_Float16)W[idx];
    }
    if (t < 32) Rms[t] = 0.0f;
    __syncthreads();

    const float* xb = x + (size_t)b * Nn * Dd;
    const float* sp = seasonal + (size_t)b * Nn * Dd;
    int r  = t >> 3;                 // 0..31
    int d0 = (t & 7) * 16;
    int n  = n0 + r;
    float ss = 0.0f;
    #pragma unroll
    for (int dd = 0; dd < 16; ++dd) {
        int d = d0 + dd;
        float xv = xb[(size_t)n * Dd + d];
        float sv = sp[(size_t)n * Dd + d];
        Th[r * Dd + d] = (_Float16)(xv - sv);          // trend (pre-blend seasonal)
        float sbl;
        if (n < BL) {
            float w1 = (float)n * (1.0f / (float)(BL - 1));
            float sm = sp[(size_t)(Nn - 1 - n) * Dd + d];
            sbl = w1 * sv + (1.0f - w1) * sm;          // head/tail blend
        } else {
            sbl = sv;
        }
        Sb[r * Dd + d] = sbl;
        ss += sbl * sbl;
    }
    atomicAdd(&Rms[r], ss);
    __syncthreads();

    int wave = t >> 5, L = t & 31;
    int ar  = L & 15;
    int klo = (L >= 16) ? 8 : 0;
    int k16 = (L >= 16) ? 16 : 0;
    int e   = wave * 16 + (L & 15);
    float la = logit_alpha[0];
    float aa = 1.0f / (1.0f + __expf(-la));
    float bv = bias[e], sc = scale[e], of = offset[e];
    int mofs = (L >= 16) ? 8 : 0;

    #pragma unroll
    for (int s = 0; s < 2; ++s) {
        v8f c = {};
        #pragma unroll
        for (int ch = 0; ch < 4; ++ch) {
            v8h lo = *(const v8h*)&Th[(s * 16 + ar) * Dd + ch * 32 + k16];
            v8h hi = *(const v8h*)&Th[(s * 16 + ar) * Dd + ch * 32 + k16 + 8];
            v16h a;
            #pragma unroll
            for (int i = 0; i < 8; ++i) { a[i] = lo[i]; a[i + 8] = hi[i]; }
            v16h bf = *(const v16h*)&Wh[e * Dd + ch * 32 + k16];
            c = __builtin_amdgcn_wmma_f32_16x16x32_f16(false, a, false, bf,
                                                       (short)0, c, false, false);
        }
        #pragma unroll
        for (int v = 0; v < 8; ++v) {
            int m = s * 16 + v + mofs;
            float proj = c[v] + bv;
            float sbl  = Sb[m * Dd + e];
            float rms  = sqrtf(Rms[m] * (1.0f / (float)Dd));   // ||row||/sqrt(D)
            float sout = sc * (sbl / (rms + 1e-8f)) + of;
            out[((size_t)b * Nn + (n0 + m)) * Dd + e] = aa * proj + (1.0f - aa) * sout;
        }
    }
}

extern "C" void kernel_launch(void* const* d_in, const int* in_sizes, int n_in,
                              void* d_out, int out_size, void* d_ws, size_t ws_size,
                              hipStream_t stream) {
    const float* x      = (const float*)d_in[0];
    const float* W      = (const float*)d_in[1];
    const float* bias   = (const float*)d_in[2];
    const float* scale  = (const float*)d_in[3];
    const float* offset = (const float*)d_in[4];
    const float* la     = (const float*)d_in[5];
    float* out = (float*)d_out;
    float* ema = out + (size_t)Bb * Nn * Dd;                 // tuple output #2

    float* grev     = (float*)d_ws;                          // 320 f32 taps
    float* eend     = (float*)((char*)d_ws + 2048);          // B*NSEG*D endpoints (128 KB)
    float* seasonal = (float*)((char*)d_ws + 2048 + (size_t)Bb * NSEG * Dd * 4);

    mofo_taps_kernel<<<1, 320, 0, stream>>>(grev);
    mofo_ema_seg_kernel<<<dim3(Bb, NSEG), Dd, 0, stream>>>(x, eend);
    mofo_ema_fill_kernel<<<dim3(Bb, NSEG), Dd, 0, stream>>>(x, eend, ema);
    mofo_conv_kernel<<<dim3(Nn / 64, 2, Bb), 128, 0, stream>>>(x, grev, seasonal);
    mofo_fuse_kernel<<<dim3(Nn / 32, Bb), 256, 0, stream>>>(x, W, bias, scale, offset,
                                                            la, seasonal, out);
}
